// MambaMIL_47691316855479
// MI455X (gfx1250) — matrix operand order repaired and checked by reading, not compiled, not run on hardware
//
#include <hip/hip_runtime.h>
#include <hip/hip_bf16.h>
#include <math.h>
#include <stdint.h>

// ---------------------------------------------------------------------------
// Types for CDNA5 WMMA (wave32): V_WMMA_F32_16X16X32_BF16
// ---------------------------------------------------------------------------
typedef __bf16 bf16;
typedef __attribute__((ext_vector_type(16))) __bf16 v16bf;
typedef __attribute__((ext_vector_type(8)))  __bf16 v8bf;
typedef __attribute__((ext_vector_type(8)))  float  v8f;
typedef int v4i_vs __attribute__((__vector_size__(4 * sizeof(int))));

#define EPI_NONE          0
#define EPI_BIAS_GELU     1
#define EPI_BIAS_SOFTPLUS 2
#define EPI_RESIDUAL      3

// gfx1250 async global->LDS copy (ASYNCcnt-tracked). Guarded so the file
// still compiles (with a sync fallback) if the builtins are absent.
#if defined(__AMDGCN__) && \
    __has_builtin(__builtin_amdgcn_global_load_async_to_lds_b128) && \
    __has_builtin(__builtin_amdgcn_s_wait_asynccnt)
#define USE_ASYNC_COPY 1
#else
#define USE_ASYNC_COPY 0
#endif

__device__ __forceinline__ float gelu_exact(float x) {
  return 0.5f * x * (1.0f + erff(x * 0.70710678118654752f));
}
__device__ __forceinline__ float softplus_f(float x) {
  return (x > 0.0f) ? (x + log1pf(expf(-x))) : log1pf(expf(x));
}
__device__ __forceinline__ float silu_f(float x) {
  return x / (1.0f + expf(-x));
}

// 16-byte global -> LDS copy (per lane)
__device__ __forceinline__ void copy16_g2lds(const bf16* __restrict__ g,
                                             bf16* __restrict__ l) {
#if USE_ASYNC_COPY
  __builtin_amdgcn_global_load_async_to_lds_b128(
      (__attribute__((address_space(1))) v4i_vs*)g,
      (__attribute__((address_space(3))) v4i_vs*)l,
      0, 0);
#else
  *reinterpret_cast<v8bf*>(l) = *reinterpret_cast<const v8bf*>(g);
#endif
}

// ---------------------------------------------------------------------------
// Tiled bf16 WMMA GEMM:  C[M,N] = epilogue( A[M,K] * W[N,K]^T )
// A, W already bf16. Block = 128 threads (4 waves). Tile 64x64, K-step 32.
// Wave w computes rows [16w,16w+16) x all 64 cols (4 accumulators).
// Optional bf16 mirror output Cb (for feeding the next bf16 GEMM).
// ---------------------------------------------------------------------------
#define BM 64
#define BN 64
#define BK 32
#define LDSA 40   // LDS row stride in bf16 elements (pad vs bank conflicts)

__global__ __launch_bounds__(128)
void gemm_bf16_wmma(const bf16* __restrict__ A, int lda,
                    const bf16* __restrict__ W, int ldw,
                    float* __restrict__ C, int ldc,
                    bf16* __restrict__ Cb, int ldcb,
                    int M, int N, int K,
                    const float* __restrict__ bias,
                    const float* __restrict__ residual, int ldr,
                    int epi)
{
  __shared__ bf16 As[BM * LDSA];
  __shared__ bf16 Bs[BN * LDSA];

  const int tid  = threadIdx.x;
  const int wave = tid >> 5;        // 0..3
  const int lane = tid & 31;
  const int l15  = lane & 15;
  const int hi16 = lane >> 4;       // 0 or 1 (half-wave)

  const int bm = blockIdx.y * BM;
  const int bn = blockIdx.x * BN;

  v8f acc[4] = {v8f{}, v8f{}, v8f{}, v8f{}};

  // fragment addressing (ISA 7.12.2, 16-bit A 16x32 / B 32x16 layouts)
  const int a_row = wave * 16 + l15;
  const int a_k0  = hi16 ? 8 : 0;      // A: halves {k0..k0+7, k0+16..k0+23}
  const int b_k0  = hi16 ? 8 : 0;      // B: halves {k0..k0+7, k0+16..k0+23}

  for (int k0 = 0; k0 < K; k0 += BK) {
    __syncthreads();
    // Tile staging: 64 rows x 32 bf16 = 256 chunks of 16B per matrix;
    // 128 threads x 2 rounds, straight global->LDS.
    #pragma unroll
    for (int r2 = 0; r2 < 2; ++r2) {
      const int ch  = tid + r2 * 128;     // 0..255
      const int row = ch >> 2;            // 0..63
      const int col = (ch & 3) * 8;       // 0,8,16,24
      copy16_g2lds(A + (size_t)(bm + row) * lda + k0 + col,
                   &As[row * LDSA + col]);
      copy16_g2lds(W + (size_t)(bn + row) * ldw + k0 + col,
                   &Bs[row * LDSA + col]);
    }
#if USE_ASYNC_COPY
    __builtin_amdgcn_s_wait_asynccnt(0);
#endif
    __syncthreads();

    union { v16bf v; v8bf h[2]; } af;
    {
      const bf16* ap = &As[a_row * LDSA + a_k0];
      af.h[0] = *reinterpret_cast<const v8bf*>(ap);        // K = k0..k0+7
      af.h[1] = *reinterpret_cast<const v8bf*>(ap + 16);   // K = k0+16..k0+23
    }
    #pragma unroll
    for (int nt = 0; nt < 4; ++nt) {
      union { v16bf v; v8bf h[2]; } bfv;
      const bf16* bp = &Bs[(nt * 16 + l15) * LDSA + b_k0];
      bfv.h[0] = *reinterpret_cast<const v8bf*>(bp);
      bfv.h[1] = *reinterpret_cast<const v8bf*>(bp + 16);
      acc[nt] = __builtin_amdgcn_wmma_f32_16x16x32_bf16(
          false, af.v, false, bfv.v, (short)0, acc[nt], false, false);
    }
  }

  // Epilogue + store. C/D layout: VGPR r -> M = r + 8*hi16; N = lane&15.
  #pragma unroll
  for (int nt = 0; nt < 4; ++nt) {
    const int n = bn + nt * 16 + l15;
    #pragma unroll
    for (int r = 0; r < 8; ++r) {
      const int m = bm + wave * 16 + hi16 * 8 + r;
      float v = acc[nt][r];
      if (epi == EPI_BIAS_GELU)          v = gelu_exact(v + bias[n]);
      else if (epi == EPI_BIAS_SOFTPLUS) v = softplus_f(v + bias[n]);
      else if (epi == EPI_RESIDUAL)      v = v + residual[(size_t)m * ldr + n];
      C[(size_t)m * ldc + n] = v;
      if (Cb) Cb[(size_t)m * ldcb + n] = (bf16)v;
    }
  }
}

// ---------------------------------------------------------------------------
// Elementwise fp32 -> bf16 conversion (weights / network input, once).
// ---------------------------------------------------------------------------
__global__ __launch_bounds__(256)
void f32_to_bf16_kernel(const float* __restrict__ in, bf16* __restrict__ out,
                        int n)
{
  const int i = blockIdx.x * 256 + threadIdx.x;
  if (i < n) out[i] = (bf16)in[i];
}

// ---------------------------------------------------------------------------
// RMSNorm over rows of length D (D=512): one block per row.
// Writes fp32 and/or bf16 output (either pointer may be null).
// ---------------------------------------------------------------------------
__global__ __launch_bounds__(256)
void rmsnorm_kernel(const float* __restrict__ x, const float* __restrict__ w,
                    float* __restrict__ outf, bf16* __restrict__ outb, int D)
{
  const int row = blockIdx.x;
  const float* xr = x + (size_t)row * D;
  float ss = 0.0f;
  for (int i = threadIdx.x; i < D; i += 256) { float v = xr[i]; ss += v * v; }
  __shared__ float red[256];
  red[threadIdx.x] = ss;
  __syncthreads();
  for (int s = 128; s > 0; s >>= 1) {
    if (threadIdx.x < s) red[threadIdx.x] += red[threadIdx.x + s];
    __syncthreads();
  }
  const float scale = rsqrtf(red[0] / (float)D + 1e-5f);
  for (int i = threadIdx.x; i < D; i += 256) {
    const float r = xr[i] * scale * w[i];
    if (outf) outf[(size_t)row * D + i] = r;
    if (outb) outb[(size_t)row * D + i] = (bf16)r;
  }
}

// ---------------------------------------------------------------------------
// Depthwise causal conv1d (K=4) + SiLU.  In: proj[:, 0:1024] (ld=projld).
// Writes fp32 (for the scan) and bf16 (for the x_proj GEMM).
// ---------------------------------------------------------------------------
__global__ __launch_bounds__(256)
void conv_silu_kernel(const float* __restrict__ proj,
                      const float* __restrict__ cw,   // [1024][4]
                      const float* __restrict__ cb,   // [1024]
                      float* __restrict__ u, bf16* __restrict__ ub,
                      int L, int Cc, int projld)
{
  const int idx = blockIdx.x * 256 + threadIdx.x;
  if (idx >= L * Cc) return;
  const int t = idx / Cc;
  const int c = idx - t * Cc;
  float acc = cb[c];
  #pragma unroll
  for (int k = 0; k < 4; ++k) {
    const int tt = t - 3 + k;
    if (tt >= 0) acc += proj[(size_t)tt * projld + c] * cw[c * 4 + k];
  }
  const float r = silu_f(acc);
  u[idx]  = r;
  ub[idx] = (bf16)r;
}

// ---------------------------------------------------------------------------
// Selective scan: thread-per-channel (1024 ch), 16-state in registers,
// B/C staged through LDS in 16-timestep chunks.
// ---------------------------------------------------------------------------
__global__ __launch_bounds__(256)
void scan_kernel(const float* __restrict__ u, const float* __restrict__ delta,
                 const float* __restrict__ dbc,       // [L][64]; B @32, C @48
                 const float* __restrict__ A_log,     // [1024][16]
                 const float* __restrict__ Dp,        // [1024]
                 float* __restrict__ y, int L)
{
  const int c = blockIdx.x * 256 + threadIdx.x;   // 0..1023
  float A[16], s[16];
  #pragma unroll
  for (int n = 0; n < 16; ++n) {
    A[n] = -expf(A_log[c * 16 + n]);
    s[n] = 0.0f;
  }
  const float dpar = Dp[c];

  __shared__ float Bs[16][16];
  __shared__ float Cs[16][16];

  for (int t0 = 0; t0 < L; t0 += 16) {
    __syncthreads();
    for (int j = threadIdx.x; j < 512; j += 256) {
      const int i = j >> 5;
      const int col = j & 31;
      const float v = dbc[(size_t)(t0 + i) * 64 + 32 + col];
      if (col < 16) Bs[i][col] = v; else Cs[i][col - 16] = v;
    }
    __syncthreads();
    #pragma unroll 4
    for (int i = 0; i < 16; ++i) {
      const int t = t0 + i;
      const float dt = delta[(size_t)t * 1024 + c];
      const float uu = u[(size_t)t * 1024 + c];
      float yy = 0.0f;
      #pragma unroll
      for (int n = 0; n < 16; ++n) {
        const float dA = expf(dt * A[n]);
        s[n] = s[n] * dA + (dt * Bs[i][n]) * uu;
        yy += s[n] * Cs[i][n];
      }
      y[(size_t)t * 1024 + c] = yy + uu * dpar;
    }
  }
}

// ---------------------------------------------------------------------------
// yb = bf16( y * silu(gate) ),  gate = proj[:, 1024 + c]
// ---------------------------------------------------------------------------
__global__ __launch_bounds__(256)
void gate_kernel(const float* __restrict__ y, const float* __restrict__ proj,
                 bf16* __restrict__ yb, int total, int projld)
{
  const int idx = blockIdx.x * 256 + threadIdx.x;
  if (idx >= total) return;
  const int t = idx >> 10;          // /1024
  const int c = idx & 1023;
  const float g = proj[(size_t)t * projld + 1024 + c];
  yb[idx] = (bf16)(y[idx] * silu_f(g));
}

// ---------------------------------------------------------------------------
// hidden[d] = mean_t hfinal[t][d]
// ---------------------------------------------------------------------------
__global__ __launch_bounds__(256)
void mean_kernel(const float* __restrict__ hf, float* __restrict__ hidden,
                 int L, int D)
{
  const int d = blockIdx.x;
  float ssum = 0.0f;
  for (int t = threadIdx.x; t < L; t += 256) ssum += hf[(size_t)t * D + d];
  __shared__ float red[256];
  red[threadIdx.x] = ssum;
  __syncthreads();
  for (int s = 128; s > 0; s >>= 1) {
    if (threadIdx.x < s) red[threadIdx.x] += red[threadIdx.x + s];
    __syncthreads();
  }
  if (threadIdx.x == 0) hidden[d] = red[0] / (float)L;
}

// ---------------------------------------------------------------------------
// logits[i] = hidden . cls_w[i] + cls_b[i]   (i = 0,1)
// ---------------------------------------------------------------------------
__global__ __launch_bounds__(256)
void logits_kernel(const float* __restrict__ hidden,
                   const float* __restrict__ cls_w,
                   const float* __restrict__ cls_b,
                   float* __restrict__ out)
{
  __shared__ float red[256];
  for (int i = 0; i < 2; ++i) {
    float ssum = 0.0f;
    for (int d = threadIdx.x; d < 512; d += 256)
      ssum += hidden[d] * cls_w[i * 512 + d];
    red[threadIdx.x] = ssum;
    __syncthreads();
    for (int s = 128; s > 0; s >>= 1) {
      if (threadIdx.x < s) red[threadIdx.x] += red[threadIdx.x + s];
      __syncthreads();
    }
    if (threadIdx.x == 0) out[i] = red[0] + cls_b[i];
    __syncthreads();
  }
}

// ---------------------------------------------------------------------------
// Host orchestration
// ---------------------------------------------------------------------------
extern "C" void kernel_launch(void* const* d_in, const int* in_sizes, int n_in,
                              void* d_out, int out_size, void* d_ws, size_t ws_size,
                              hipStream_t stream)
{
  (void)in_sizes; (void)n_in; (void)out_size; (void)ws_size;

  constexpr int L = 2048, D_IN = 1024, DM = 512, DI = 1024;

  const float* x         = (const float*)d_in[0];   // [L, D_IN]
  const float* W1        = (const float*)d_in[1];   // [DM, D_IN]
  const float* b1        = (const float*)d_in[2];   // [DM]
  const float* norm_w    = (const float*)d_in[3];   // [2, DM]
  const float* in_proj_w = (const float*)d_in[4];   // [2, 2*DI, DM]
  const float* conv_w    = (const float*)d_in[5];   // [2, DI, 4]
  const float* conv_b    = (const float*)d_in[6];   // [2, DI]
  const float* x_proj_w  = (const float*)d_in[7];   // [2, 64, DI]
  const float* dt_proj_w = (const float*)d_in[8];   // [2, DI, 32]
  const float* dt_proj_b = (const float*)d_in[9];   // [2, DI]
  const float* A_log     = (const float*)d_in[10];  // [2, DI, 16]
  const float* Dp        = (const float*)d_in[11];  // [2, DI]
  const float* out_proj_w= (const float*)d_in[12];  // [2, DM, DI]
  const float* norm_f_w  = (const float*)d_in[13];  // [DM]
  const float* cls_w     = (const float*)d_in[14];  // [2, DM]
  const float* cls_b     = (const float*)d_in[15];  // [2]

  // fp32 workspace
  float* f     = (float*)d_ws;
  float* h     = f;                          // [L, DM]
  float* proj  = h     + (size_t)L * DM;     // [L, 2*DI]
  float* u     = proj  + (size_t)L * 2 * DI; // [L, DI]
  float* dbc   = u     + (size_t)L * DI;     // [L, 64]
  float* delta = dbc   + (size_t)L * 64;     // [L, DI]
  float* y     = delta + (size_t)L * DI;     // [L, DI]

  // bf16 workspace (operands for the WMMA GEMMs)
  bf16* b     = (bf16*)(y + (size_t)L * DI);
  bf16* xb    = b;                                 // [L, D_IN]
  bf16* w1b   = xb    + (size_t)L * D_IN;          // [DM, D_IN]
  bf16* inpb  = w1b   + (size_t)DM * D_IN;         // [2, 2*DI, DM]
  bf16* xpb   = inpb  + (size_t)2 * 2 * DI * DM;   // [2, 64, DI]
  bf16* dtpb  = xpb   + (size_t)2 * 64 * DI;       // [2, DI, 32]
  bf16* outpb = dtpb  + (size_t)2 * DI * 32;       // [2, DM, DI]
  bf16* hsb   = outpb + (size_t)2 * DM * DI;       // [L, DM]
  bf16* ub    = hsb   + (size_t)L * DM;            // [L, DI]
  bf16* dbcb  = ub    + (size_t)L * DI;            // [L, 64]
  bf16* yb    = dbcb  + (size_t)L * 64;            // [L, DI]

  float* outf   = (float*)d_out;
  float* logits = outf;             // [2]
  float* hidden = outf + 2;         // [DM]
  float* hfinal = outf + 2 + DM;    // [L, DM]

  const dim3 blk(128);
  auto cvt = [&](const float* src, bf16* dst, int n) {
    f32_to_bf16_kernel<<<(n + 255) / 256, 256, 0, stream>>>(src, dst, n);
  };

  // One-time bf16 conversion of GEMM operands (input + all weights)
  cvt(x,          xb,    L * D_IN);
  cvt(W1,         w1b,   DM * D_IN);
  cvt(in_proj_w,  inpb,  2 * 2 * DI * DM);
  cvt(x_proj_w,   xpb,   2 * 64 * DI);
  cvt(dt_proj_w,  dtpb,  2 * DI * 32);
  cvt(out_proj_w, outpb, 2 * DM * DI);

  // h = gelu(x @ W1^T + b1)
  gemm_bf16_wmma<<<dim3(DM / BN, L / BM), blk, 0, stream>>>(
      xb, D_IN, w1b, D_IN, h, DM, nullptr, 0,
      L, DM, D_IN, b1, nullptr, 0, EPI_BIAS_GELU);

  for (int l = 0; l < 2; ++l) {
    // hsb = bf16(rmsnorm(h))
    rmsnorm_kernel<<<L, 256, 0, stream>>>(h, norm_w + (size_t)l * DM,
                                          nullptr, hsb, DM);
    // proj = hs @ in_proj_w^T  [L, 2*DI]
    gemm_bf16_wmma<<<dim3(2 * DI / BN, L / BM), blk, 0, stream>>>(
        hsb, DM, inpb + (size_t)l * 2 * DI * DM, DM,
        proj, 2 * DI, nullptr, 0,
        L, 2 * DI, DM, nullptr, nullptr, 0, EPI_NONE);
    // u/ub = silu(causal_conv1d(proj[:, :DI]))
    conv_silu_kernel<<<(L * DI + 255) / 256, 256, 0, stream>>>(
        proj, conv_w + (size_t)l * DI * 4, conv_b + (size_t)l * DI,
        u, ub, L, DI, 2 * DI);
    // dbc = u @ x_proj_w^T  [L, 64]   (+ bf16 mirror for the dt GEMM)
    gemm_bf16_wmma<<<dim3(64 / BN, L / BM), blk, 0, stream>>>(
        ub, DI, xpb + (size_t)l * 64 * DI, DI,
        dbc, 64, dbcb, 64,
        L, 64, DI, nullptr, nullptr, 0, EPI_NONE);
    // delta = softplus(dbc[:, :32] @ dt_proj_w^T + dt_proj_b)  [L, DI]
    gemm_bf16_wmma<<<dim3(DI / BN, L / BM), blk, 0, stream>>>(
        dbcb, 64, dtpb + (size_t)l * DI * 32, 32,
        delta, DI, nullptr, 0,
        L, DI, 32, dt_proj_b + (size_t)l * DI, nullptr, 0, EPI_BIAS_SOFTPLUS);
    // y = selective_scan(u, delta, A, B, C, Dp)
    scan_kernel<<<DI / 256, 256, 0, stream>>>(
        u, delta, dbc, A_log + (size_t)l * DI * 16, Dp + (size_t)l * DI, y, L);
    // yb = bf16(y * silu(gate))
    gate_kernel<<<(L * DI + 255) / 256, 256, 0, stream>>>(
        y, proj, yb, L * DI, 2 * DI);
    // h = h + y @ out_proj_w^T
    gemm_bf16_wmma<<<dim3(DM / BN, L / BM), blk, 0, stream>>>(
        yb, DI, outpb + (size_t)l * DM * DI, DI,
        h, DM, nullptr, 0,
        L, DM, DI, nullptr, h, DM, EPI_RESIDUAL);
  }

  // hfinal = rmsnorm(h, norm_f_w)  -> third output
  rmsnorm_kernel<<<L, 256, 0, stream>>>(h, norm_f_w, hfinal, nullptr, DM);
  // hidden = mean_t hfinal        -> second output
  mean_kernel<<<DM, 256, 0, stream>>>(hfinal, hidden, L, DM);
  // logits                         -> first output
  logits_kernel<<<1, 256, 0, stream>>>(hidden, cls_w, cls_b, logits);
}
